// MaskedCausalAttention_57939108823715
// MI455X (gfx1250) — matrix operand order repaired and checked by reading, not compiled, not run on hardware
//
#include <hip/hip_runtime.h>

// ---------------------------------------------------------------------------
// MaskedCausalAttention for MI455X (gfx1250), wave32 + WMMA bf16 + TDM.
//   B=4, T=2048, C=1024, N=16 heads, D=64.
// Pipeline:
//   1) convert x and the 4 weight matrices fp32 -> bf16 (workspace)
//   2) Q/K/V GEMMs (x @ W^T + b) via v_wmma_f32_16x16x32_bf16; A-tile staged
//      into LDS by the Tensor Data Mover (double buffered, s_wait_tensorcnt)
//   3) flash attention (online softmax, causal + token mask); K tile via TDM
//   4) output projection GEMM -> fp32 d_out
// ---------------------------------------------------------------------------

typedef __bf16 bf16;
typedef __attribute__((ext_vector_type(4)))  bf16  v4bf;
typedef __attribute__((ext_vector_type(8)))  bf16  v8bf;
typedef __attribute__((ext_vector_type(16))) bf16  v16bf;
typedef __attribute__((ext_vector_type(4)))  float v4f;
typedef __attribute__((ext_vector_type(8)))  float v8f;

constexpr int Bz = 4, Tt = 2048, Cc = 1024, NH = 16, HD = 64;
constexpr int Mrows = Bz * Tt;           // 8192
constexpr int BM = 128, BN = 128, BK = 32;
constexpr int LDA = BK + 8;              // 40 elems = 80 B row stride (16B-aligned)

// ---- Tensor Data Mover support probes (device pass only) -------------------
#if defined(__AMDGCN__)
#  define USE_TDM 1
#  if __has_builtin(__builtin_amdgcn_tensor_load_to_lds)
#    warning "probe: __builtin_amdgcn_tensor_load_to_lds PRESENT"
#    define TDM_BUILTIN 1
#  else
#    warning "probe: __builtin_amdgcn_tensor_load_to_lds ABSENT -> inline asm"
#    define TDM_BUILTIN 0
#  endif
#  if __has_builtin(__builtin_amdgcn_s_wait_tensorcnt)
#    warning "probe: __builtin_amdgcn_s_wait_tensorcnt PRESENT"
#    define TCNT_BUILTIN 1
#  else
#    warning "probe: __builtin_amdgcn_s_wait_tensorcnt ABSENT -> inline asm"
#    define TCNT_BUILTIN 0
#  endif
#  if __clang_major__ >= 23
#    warning "probe: clang major >= 23 (try 6-arg TDM builtin)"
#    define TDM_6ARG 1
#  else
#    warning "probe: clang major <= 22 (try 5-arg TDM builtin)"
#    define TDM_6ARG 0
#  endif
#else
#  define USE_TDM 0
#endif

#if USE_TDM
typedef __attribute__((ext_vector_type(4))) unsigned int v4u;
typedef __attribute__((ext_vector_type(8))) int          v8i;
typedef __attribute__((ext_vector_type(4))) int          v4i;

__device__ __forceinline__ void tdm_issue(v4u g0, v8i g1) {
  const v4i z = {0, 0, 0, 0};
#if TDM_BUILTIN
#if TDM_6ARG
  const v8i z8 = {0, 0, 0, 0, 0, 0, 0, 0};
  __builtin_amdgcn_tensor_load_to_lds(g0, g1, z, z, z8, 0);
#else
  __builtin_amdgcn_tensor_load_to_lds(g0, g1, z, z, 0);
#endif
#else
  // VIMAGE TENSOR_LOAD_TO_LDS: D# groups in SGPR tuples, groups 2/3 zeroed
  asm volatile("tensor_load_to_lds %0, %1, %2, %3"
               :
               : "s"(g0), "s"(g1), "s"(z), "s"(z)
               : "memory");
#endif
}

__device__ __forceinline__ void tdm_wait0() {
#if TCNT_BUILTIN
  __builtin_amdgcn_s_wait_tensorcnt((short)0);
#else
  asm volatile("s_wait_tensorcnt 0x0" ::: "memory");
#endif
}

// Issue a 2-D tensor_load_to_lds per ISA ch.8 D# layout.
//   tile_d0/tile_d1/tensor_d0/tensor_d1/stride_d0 in bf16 elements.
//   pad_icode: dwords-before-pad = 2<<pad_icode ; pad_acode: pad = pad_acode+1 dwords
__device__ __forceinline__ void tdm_load_2d(const void* gaddr, unsigned lds_addr,
                                            unsigned tile_d0, unsigned tile_d1,
                                            unsigned tensor_d0, unsigned tensor_d1,
                                            unsigned stride_d0,
                                            unsigned pad_icode, unsigned pad_acode) {
  const unsigned long long ga = (unsigned long long)(uintptr_t)gaddr;
  v4u g0;
  g0[0] = 1u;                                        // count=1, user descriptor
  g0[1] = lds_addr;                                  // lds_addr [63:32]
  g0[2] = (unsigned)(ga & 0xFFFFFFFFu);              // global_addr[31:0]
  g0[3] = (unsigned)((ga >> 32) & 0x1FFFFFFu)        // global_addr[56:32]
          | 0x80000000u;                             // type=2 ("image") [127:126]
  v8i g1;
  g1[0] = (int)((1u << 16)                           // data_size = 2 bytes
                | (1u << 20)                         // pad_enable
                | (pad_icode << 22)                  // pad_interval
                | (pad_acode << 25));                // pad_amount
  g1[1] = (int)((tensor_d0 & 0xFFFFu) << 16);        // tensor_dim0[15:0] @ [63:48]
  g1[2] = (int)(((tensor_d0 >> 16) & 0xFFFFu)        // tensor_dim0[31:16]
                | ((tensor_d1 & 0xFFFFu) << 16));    // tensor_dim1[15:0]
  g1[3] = (int)(((tensor_d1 >> 16) & 0xFFFFu)        // tensor_dim1[31:16]
                | ((tile_d0 & 0xFFFFu) << 16));      // tile_dim0 @ [127:112]
  g1[4] = (int)(tile_d1 & 0xFFFFu);                  // tile_dim1 ; tile_dim2 = 0
  g1[5] = (int)stride_d0;                            // tensor_dim0_stride[31:0]
  g1[6] = 0;                                         // stride[47:32], dim1_stride
  g1[7] = 0;
  tdm_issue(g0, g1);
}
#endif  // USE_TDM

__device__ __forceinline__ v16bf join16(v8bf a, v8bf b) {
  return __builtin_shufflevector(a, b, 0, 1, 2, 3, 4, 5, 6, 7,
                                 8, 9, 10, 11, 12, 13, 14, 15);
}

// ---------------------------------------------------------------------------
// fp32 -> bf16 conversion (4 floats / thread)
// ---------------------------------------------------------------------------
__global__ __launch_bounds__(256) void cvt_f32_bf16(const float* __restrict__ src,
                                                    bf16* __restrict__ dst, int n4) {
  int i = blockIdx.x * blockDim.x + threadIdx.x;
  if (i < n4) {
    v4f f = ((const v4f*)src)[i];
    v4bf o;
    o[0] = (bf16)f[0]; o[1] = (bf16)f[1]; o[2] = (bf16)f[2]; o[3] = (bf16)f[3];
    ((v4bf*)dst)[i] = o;
  }
}

// ---------------------------------------------------------------------------
// GEMM: out(M,Cc) = A(M,Cc) @ W^T + bias.  B-matrix column j == W row j.
// 8 waves; WG tile 128x128; wave tile 64x32.  A tile staged in LDS by TDM
// (double buffered) -- wave 0 drives the DMA, everyone else just computes.
// mode 0: bf16 head-major (b,h,t,d);  mode 1: fp32 row-major (M,Cc)
// ---------------------------------------------------------------------------
__global__ __launch_bounds__(256) void gemm_bf16_kernel(
    const bf16* __restrict__ A, const bf16* __restrict__ W,
    const float* __restrict__ bias, bf16* __restrict__ outH,
    float* __restrict__ outF, int mode) {
  __shared__ bf16 As[2][BM * LDA];

  const int bm = blockIdx.x * BM;
  const int bn = blockIdx.y * BN;
  const int tid = threadIdx.x;
  const int lane = tid & 31;
  const int wid = tid >> 5;
  const int l16 = lane & 15, lhalf = lane >> 4;
  const int wm = (wid >> 2) * 64;   // 0 / 64
  const int wn = (wid & 3) * 32;    // 0 / 32 / 64 / 96

  v8f acc[4][2] = {};

#if USE_TDM
  // prologue: kick off DMA of tile 0.  Rows are 32 bf16 = 16 dwords, pad 4
  // dwords -> 80 B LDS row stride == LDA.
  if (wid == 0)
    tdm_load_2d(A + (size_t)bm * Cc, (unsigned)(uintptr_t)&As[0][0],
                BK, BM, Cc, Mrows, Cc, /*pad_i=*/3, /*pad_a=*/3);
#endif

  for (int kt = 0; kt < Cc / BK; ++kt) {
    const int k0 = kt * BK;
    const bf16* Asb = &As[kt & 1][0];

#if USE_TDM
    if (wid == 0) tdm_wait0();
    __syncthreads();  // tile kt visible to all; prev compute done w/ other buf
    if (wid == 0 && k0 + BK < Cc)
      tdm_load_2d(A + (size_t)bm * Cc + (k0 + BK),
                  (unsigned)(uintptr_t)&As[(kt + 1) & 1][0],
                  BK, BM, Cc, Mrows, Cc, 3, 3);
#else
    {  // manual staging fallback (also keeps host pass happy)
      const int row = tid >> 1;
      const int col = (tid & 1) * 16;
      const bf16* src = A + (size_t)(bm + row) * Cc + (k0 + col);
      v8bf t0 = *(const v8bf*)src;
      v8bf t1 = *(const v8bf*)(src + 8);
      *(v8bf*)&As[kt & 1][row * LDA + col] = t0;
      *(v8bf*)&As[kt & 1][row * LDA + col + 8] = t1;
    }
    __syncthreads();
#endif

    // ---- B fragments straight from global (L2-resident weights) -------
    v16bf bfr[2];
#pragma unroll
    for (int s = 0; s < 2; ++s) {
      const int col = bn + wn + s * 16 + l16;                 // output column
      const bf16* wp = W + (size_t)col * Cc + k0 + lhalf * 8;
      bfr[s] = join16(*(const v8bf*)wp, *(const v8bf*)(wp + 16));
      if (k0 + BK < Cc) __builtin_prefetch(wp + BK, 0, 2);
    }
    // ---- A fragments from LDS + WMMA -----------------------------------
#pragma unroll
    for (int i = 0; i < 4; ++i) {
      const bf16* ap = &Asb[(wm + i * 16 + l16) * LDA + lhalf * 8];
      v16bf af = join16(*(const v8bf*)ap, *(const v8bf*)(ap + 16));
#pragma unroll
      for (int s = 0; s < 2; ++s)
        acc[i][s] = __builtin_amdgcn_wmma_f32_16x16x32_bf16(
            false, af, false, bfr[s], (short)0, acc[i][s], false, false);
    }
#if !USE_TDM
    __syncthreads();
#endif
  }

  // ---- epilogue: bias, convert, scatter ---------------------------------
#pragma unroll
  for (int i = 0; i < 4; ++i) {
#pragma unroll
    for (int s = 0; s < 2; ++s) {
      const int col = bn + wn + s * 16 + l16;
      const float bb = bias[col];
#pragma unroll
      for (int r = 0; r < 8; ++r) {
        const int m = bm + wm + i * 16 + r + 8 * lhalf;
        const float val = acc[i][s][r] + bb;
        if (mode == 0) {
          const int bIdx = m >> 11, t = m & (Tt - 1);     // T = 2048
          const int h = col >> 6, d = col & (HD - 1);     // D = 64
          outH[(((size_t)bIdx * NH + h) * Tt + t) * HD + d] = (bf16)val;
        } else {
          outF[(size_t)m * Cc + col] = val;
        }
      }
    }
  }
}

// ---------------------------------------------------------------------------
// Flash attention.  One WG (128 thr = 4 waves) per (b, h, 64-q-row block).
// K tile DMA'd by TDM into padded LDS; V tile transposed manually (TDM has
// no transpose); P re-laid out via per-wave LDS buffer.  Online softmax.
// ---------------------------------------------------------------------------
__global__ __launch_bounds__(128) void flash_attn_kernel(
    const bf16* __restrict__ Q, const bf16* __restrict__ K,
    const bf16* __restrict__ V, const int* __restrict__ token_mask,
    bf16* __restrict__ Out) {
  __shared__ bf16 Ks[32 * 72];     // [key][d], d padded 64->72 (144 B rows)
  __shared__ bf16 Vt[HD * 40];     // [d][key], key padded 32->40
  __shared__ bf16 Ps[4][16 * 32];  // per-wave P scratch (q x key)

  const int nq = Tt / 64;                      // 32 q-blocks per (b,h)
  const int qblk = blockIdx.x % nq;
  const int bh = blockIdx.x / nq;
  const int b = bh / NH, h = bh % NH;

  const size_t hb = ((size_t)b * NH + h) * Tt * HD;
  const bf16* Qh = Q + hb;
  const bf16* Kh = K + hb;
  const bf16* Vh = V + hb;

  const int tid = threadIdx.x;
  const int lane = tid & 31, wid = tid >> 5;
  const int l16 = lane & 15, lhalf = lane >> 4;
  const int q0 = qblk * 64 + wid * 16;         // this wave's first q row

  // Q fragments (16 x 64, two k-chunks of 32)
  v16bf qf[2];
#pragma unroll
  for (int c = 0; c < 2; ++c) {
    const bf16* qp = Qh + (size_t)(q0 + l16) * HD + c * 32 + lhalf * 8;
    qf[c] = join16(*(const v8bf*)qp, *(const v8bf*)(qp + 16));
  }

  v8f o[4] = {};
  float mrow[8], lrow[8];
#pragma unroll
  for (int r = 0; r < 8; ++r) { mrow[r] = -3.0e38f; lrow[r] = 0.0f; }

  const int ntiles = (qblk + 1) * 2;           // causal: keys < qblk*64+64
  for (int kt = 0; kt < ntiles; ++kt) {
    const int kk0 = kt * 32;
    // ---- stage K (TDM) and transposed V (manual) ------------------------
#if USE_TDM
    if (wid == 0)  // K rows: 64 bf16 = 32 dwords, pad 4 dwords -> 144 B stride
      tdm_load_2d(Kh + (size_t)kk0 * HD, (unsigned)(uintptr_t)&Ks[0],
                  HD, 32, HD, Tt, HD, /*pad_i=*/4, /*pad_a=*/3);
#endif
    {
      const int row = tid >> 2;                // key 0..31
      const int col = (tid & 3) * 16;          // d chunk
#if !USE_TDM
      const bf16* kp = Kh + (size_t)(kk0 + row) * HD + col;
      v8bf k0v = *(const v8bf*)kp;
      v8bf k1v = *(const v8bf*)(kp + 8);
      *(v8bf*)&Ks[row * 72 + col] = k0v;
      *(v8bf*)&Ks[row * 72 + col + 8] = k1v;
#endif
      const bf16* vp = Vh + (size_t)(kk0 + row) * HD + col;
      v8bf v0v = *(const v8bf*)vp;
      v8bf v1v = *(const v8bf*)(vp + 8);
#pragma unroll
      for (int e = 0; e < 8; ++e) {
        Vt[(col + e) * 40 + row] = v0v[e];
        Vt[(col + 8 + e) * 40 + row] = v1v[e];
      }
      if (kt + 1 < ntiles) __builtin_prefetch(vp + 32 * HD, 0, 2);
    }
#if USE_TDM
    if (wid == 0) tdm_wait0();
#endif
    __syncthreads();

    // token-mask bits for this lane's two key columns
    int tmask0 = token_mask[(size_t)b * Tt + kk0 + l16];
    int tmask1 = token_mask[(size_t)b * Tt + kk0 + 16 + l16];

    // ---- S = Q K^T (two 16-key subtiles, two d-chunks each) ------------
    v8f sacc[2] = {};
#pragma unroll
    for (int s = 0; s < 2; ++s) {
#pragma unroll
      for (int c = 0; c < 2; ++c) {
        const bf16* kb = &Ks[(s * 16 + l16) * 72 + c * 32 + lhalf * 8];
        v16bf kf = join16(*(const v8bf*)kb, *(const v8bf*)(kb + 16));
        sacc[s] = __builtin_amdgcn_wmma_f32_16x16x32_bf16(
            false, qf[c], false, kf, (short)0, sacc[s], false, false);
      }
    }

    // ---- mask, online softmax ------------------------------------------
    float p0v[8], p1v[8], corr[8];
#pragma unroll
    for (int r = 0; r < 8; ++r) {
      const int qrow = q0 + r + 8 * lhalf;
      float s0 = sacc[0][r] * 0.125f;                  // 1/sqrt(64)
      float s1 = sacc[1][r] * 0.125f;
      if (!((kk0 + l16) <= qrow && tmask0 != 0)) s0 = -3.0e38f;
      if (!((kk0 + 16 + l16) <= qrow && tmask1 != 0)) s1 = -3.0e38f;
      float mx = fmaxf(s0, s1);
#pragma unroll
      for (int x = 1; x <= 8; x <<= 1) mx = fmaxf(mx, __shfl_xor(mx, x, 32));
      const float nm = fmaxf(mrow[r], mx);
      const float c0 = __expf(mrow[r] - nm);
      const float p0 = __expf(s0 - nm);
      const float p1 = __expf(s1 - nm);
      float rs = p0 + p1;
#pragma unroll
      for (int x = 1; x <= 8; x <<= 1) rs += __shfl_xor(rs, x, 32);
      lrow[r] = lrow[r] * c0 + rs;
      mrow[r] = nm;
      corr[r] = c0;
      p0v[r] = p0;
      p1v[r] = p1;
    }
#pragma unroll
    for (int f = 0; f < 4; ++f)
#pragma unroll
      for (int r = 0; r < 8; ++r) o[f][r] *= corr[r];

    // ---- P -> LDS -> A-fragment layout ---------------------------------
#pragma unroll
    for (int r = 0; r < 8; ++r) {
      Ps[wid][(r + 8 * lhalf) * 32 + l16] = (bf16)p0v[r];
      Ps[wid][(r + 8 * lhalf) * 32 + 16 + l16] = (bf16)p1v[r];
    }
    const bf16* pp = &Ps[wid][l16 * 32 + lhalf * 8];
    v16bf pf = join16(*(const v8bf*)pp, *(const v8bf*)(pp + 16));

    // ---- O += P V (four 16-d subtiles) ---------------------------------
#pragma unroll
    for (int f = 0; f < 4; ++f) {
      const bf16* vb = &Vt[(f * 16 + l16) * 40 + lhalf * 8];
      v16bf vf = join16(*(const v8bf*)vb, *(const v8bf*)(vb + 16));
      o[f] = __builtin_amdgcn_wmma_f32_16x16x32_bf16(
          false, pf, false, vf, (short)0, o[f], false, false);
    }
    __syncthreads();
  }

  // ---- normalize and store (b, t, c) bf16 -------------------------------
  float rinv[8];
#pragma unroll
  for (int r = 0; r < 8; ++r) rinv[r] = 1.0f / lrow[r];
#pragma unroll
  for (int f = 0; f < 4; ++f)
#pragma unroll
    for (int r = 0; r < 8; ++r) {
      const int t = q0 + r + 8 * lhalf;
      const int cidx = h * HD + f * 16 + l16;
      Out[((size_t)b * Tt + t) * Cc + cidx] = (bf16)(o[f][r] * rinv[r]);
    }
}

// ---------------------------------------------------------------------------
extern "C" void kernel_launch(void* const* d_in, const int* in_sizes, int n_in,
                              void* d_out, int out_size, void* d_ws, size_t ws_size,
                              hipStream_t stream) {
  (void)in_sizes; (void)n_in; (void)out_size; (void)ws_size;
  const float* x = (const float*)d_in[0];
  const int* tmask = (const int*)d_in[1];
  const float* wq = (const float*)d_in[2];
  const float* bq = (const float*)d_in[3];
  const float* wk = (const float*)d_in[4];
  const float* bk = (const float*)d_in[5];
  const float* wv = (const float*)d_in[6];
  const float* bv = (const float*)d_in[7];
  const float* wp = (const float*)d_in[8];
  const float* bp = (const float*)d_in[9];
  float* out = (float*)d_out;

  // workspace carve (bf16): x, 4 weights, q/k/v (head-major), attn (b,t,c)
  const size_t nXC = (size_t)Bz * Tt * Cc;  // 8,388,608
  const size_t nW = (size_t)Cc * Cc;        // 1,048,576
  bf16* xb = (bf16*)d_ws;
  bf16* wqb = xb + nXC;
  bf16* wkb = wqb + nW;
  bf16* wvb = wkb + nW;
  bf16* wpb = wvb + nW;
  bf16* qh = wpb + nW;
  bf16* kh = qh + nXC;
  bf16* vh = kh + nXC;
  bf16* ab = vh + nXC;   // ~92 MB total

  const int thr = 256;
  cvt_f32_bf16<<<dim3((unsigned)((nXC / 4 + thr - 1) / thr)), thr, 0, stream>>>(x, xb, (int)(nXC / 4));
  cvt_f32_bf16<<<dim3((unsigned)((nW / 4 + thr - 1) / thr)), thr, 0, stream>>>(wq, wqb, (int)(nW / 4));
  cvt_f32_bf16<<<dim3((unsigned)((nW / 4 + thr - 1) / thr)), thr, 0, stream>>>(wk, wkb, (int)(nW / 4));
  cvt_f32_bf16<<<dim3((unsigned)((nW / 4 + thr - 1) / thr)), thr, 0, stream>>>(wv, wvb, (int)(nW / 4));
  cvt_f32_bf16<<<dim3((unsigned)((nW / 4 + thr - 1) / thr)), thr, 0, stream>>>(wp, wpb, (int)(nW / 4));

  dim3 gg(Mrows / BM, Cc / BN);  // 64 x 8
  gemm_bf16_kernel<<<gg, 256, 0, stream>>>(xb, wqb, bq, qh, nullptr, 0);
  gemm_bf16_kernel<<<gg, 256, 0, stream>>>(xb, wkb, bk, kh, nullptr, 0);
  gemm_bf16_kernel<<<gg, 256, 0, stream>>>(xb, wvb, bv, vh, nullptr, 0);

  flash_attn_kernel<<<dim3(Bz * NH * (Tt / 64)), 128, 0, stream>>>(qh, kh, vh, tmask, ab);

  gemm_bf16_kernel<<<gg, 256, 0, stream>>>(ab, wpb, bp, nullptr, out, 1);
}
